// E_GAT_71725953843750
// MI455X (gfx1250) — compile-verified, matrix operand order checked by reading, not run on hardware
//
#include <hip/hip_runtime.h>
#include <hip/hip_bf16.h>

// ---------------------------------------------------------------------------
// EGNN layer for MI455X (gfx1250, wave32, WMMA 16x16x32 f16->f32)
//   N=10000 nodes, E=320000 edges, D=128, C=3
//   Edge kernel: 64 edges/block (4 M-tiles), 8 waves each owning 16 N-columns;
//   B fragments held in registers across the 4 M-tiles.
// ---------------------------------------------------------------------------

#define NN 10000
#define EE 320000
#define DD 128
#define MT 4          // M-tiles per edge block
#define EPB 64        // edges per block = MT*16

typedef __attribute__((ext_vector_type(16))) _Float16 v16h;
typedef __attribute__((ext_vector_type(8)))  float    v8f;

struct alignas(16) U128 { unsigned int x, y, z, w; };
struct U256 { U128 lo, hi; };

static __device__ __forceinline__ v16h frag_from(U128 lo, U128 hi) {
    U256 t; t.lo = lo; t.hi = hi;
    v16h v;
    __builtin_memcpy(&v, &t, sizeof(v));
    return v;
}

// A-fragment (16xK tile, f16) from LDS. Per ISA 16-bit A 16x32 layout:
// lane L: M = L&15; K runs of 8 at base kc*32 + (L>=16 ? 8 : 0) and +16.
static __device__ __forceinline__ v16h load_a_lds(const _Float16* base, int strideH,
                                                  int kc, int lane) {
    const _Float16* p = base + (lane & 15) * strideH + (kc << 5) + ((lane & 16) ? 8 : 0);
    return frag_from(*(const U128*)p, *(const U128*)(p + 16));
}

// B-fragment from pre-swizzled global buffer: 16 contiguous halves per lane.
static __device__ __forceinline__ v16h load_b_frag(const _Float16* frag, int kcCount,
                                                   int ntile, int kc, int lane) {
    const _Float16* p = frag + ((size_t)((ntile * kcCount + kc) * 32 + lane)) * 16;
    return frag_from(*(const U128*)p, *(const U128*)(p + 8));
}

static __device__ __forceinline__ v8f wmma_f16(v16h a, v16h b, v8f c) {
    return __builtin_amdgcn_wmma_f32_16x16x32_f16(false, a, false, b, (short)0, c,
                                                  false, false);
}

static __device__ __forceinline__ float sigmoidf_(float x) {
    return 1.0f / (1.0f + __expf(-x));
}
static __device__ __forceinline__ float siluf_(float x) { return x * sigmoidf_(x); }

// 16-lane butterfly sum (keeps lanes 0-15 and 16-31 independent; wave32).
static __device__ __forceinline__ float reduce16(float v) {
    v += __shfl_xor(v, 1, 32);
    v += __shfl_xor(v, 2, 32);
    v += __shfl_xor(v, 4, 32);
    v += __shfl_xor(v, 8, 32);
    return v;
}

// ---------------------------------------------------------------------------
// Prep kernels
// ---------------------------------------------------------------------------

// f32 -> f16 row-major copy (for the h gather path)
__global__ void cvt_f16_kernel(const float* __restrict__ src, _Float16* __restrict__ dst,
                               int nElems) {
    int i = blockIdx.x * blockDim.x + threadIdx.x;
    if (i < nElems) dst[i] = (_Float16)src[i];
}

// f32 (KxN row-major, N=128) -> f16 WMMA B-fragment order.
// dst[((ntile*KC + kc)*32 + lane)*16 + e]
//   n = ntile*16 + (lane&15)
//   kb = kc*32 + (lane&16 ? 8 : 0);  k = e<8 ? kb+e : kb+16+(e-8)
__global__ void prep_frag_kernel(const float* __restrict__ src, _Float16* __restrict__ dst,
                                 int kcCount) {
    int i = blockIdx.x * blockDim.x + threadIdx.x;
    int total = 8 * kcCount * 512;
    if (i >= total) return;
    int e     = i & 15;
    int lane  = (i >> 4) & 31;
    int kc    = (i >> 9) % kcCount;
    int ntile = i / (512 * kcCount);
    int n  = ntile * 16 + (lane & 15);
    int kb = kc * 32 + ((lane & 16) ? 8 : 0);
    int k  = (e < 8) ? (kb + e) : (kb + 16 + (e - 8));
    dst[i] = (_Float16)src[(size_t)k * DD + n];
}

__global__ void zero_kernel(float* __restrict__ p, int n) {
    int i = blockIdx.x * blockDim.x + threadIdx.x;
    if (i < n) p[i] = 0.0f;
}

// ---------------------------------------------------------------------------
// Edge kernel: 64 edges / block, 8 waves; B fragments reused across 4 M-tiles
// ---------------------------------------------------------------------------
__global__ __launch_bounds__(256) void egnn_edge_kernel(
    const _Float16* __restrict__ hf16, const int* __restrict__ erow,
    const int* __restrict__ ecol, const float* __restrict__ coord,
    const float* __restrict__ We1, const float* __restrict__ be1,
    const float* __restrict__ be2, const float* __restrict__ Wa,
    const float* __restrict__ ba,  const float* __restrict__ bc1,
    const float* __restrict__ Wc2,
    const _Float16* __restrict__ we1h, const _Float16* __restrict__ we2h,
    const _Float16* __restrict__ wc1h,
    float* __restrict__ agg_h, float* __restrict__ agg_c, float* __restrict__ cnt)
{
    __shared__ __align__(16) _Float16 ein[EPB * 256];   // [h[row] | h[col]] f16 (32 KB)
    __shared__ __align__(16) _Float16 m1s[EPB * 128];   // silu(e_in@We1+b)    (16 KB)
    __shared__ __align__(16) _Float16 m2s[EPB * 128];   // m = silu(..@We2+b)  (16 KB)
    __shared__ int   rows_s[EPB];
    __shared__ int   cols_s[EPB];
    __shared__ float diff_s[EPB][3];
    __shared__ float rad_s[EPB];
    __shared__ float satt[EPB];
    __shared__ float sw_s[EPB];

    const int tid  = threadIdx.x;
    const int lane = tid & 31;
    const int wave = tid >> 5;
    const int e0   = blockIdx.x * EPB;

    if (tid == 0) __builtin_prefetch(we1h, 0, 1);      // global_prefetch_b8

    if (tid < EPB) {
        int e = e0 + tid;
        int r = erow[e], c = ecol[e];
        rows_s[tid] = r;
        cols_s[tid] = c;
        float dx = coord[r * 3 + 0] - coord[c * 3 + 0];
        float dy = coord[r * 3 + 1] - coord[c * 3 + 1];
        float dz = coord[r * 3 + 2] - coord[c * 3 + 2];
        diff_s[tid][0] = dx; diff_s[tid][1] = dy; diff_s[tid][2] = dz;
        rad_s[tid] = dx * dx + dy * dy + dz * dz;
        satt[tid] = 0.0f;
        sw_s[tid] = 0.0f;
    }
    __syncthreads();

    // -------- gather e_in = [h[row] | h[col]] (f16, 16B vector copies) --------
    {
        int m0 = tid >> 4;
        int cb = (tid & 15) << 3;
#pragma unroll
        for (int it = 0; it < MT; ++it) {
            int m = m0 + (it << 4);
            *(U128*)(ein + m * 256 + cb) =
                *(const U128*)(hf16 + (size_t)rows_s[m] * DD + cb);
            *(U128*)(ein + m * 256 + 128 + cb) =
                *(const U128*)(hf16 + (size_t)cols_s[m] * DD + cb);
        }
    }
    __syncthreads();

    const int n   = (wave << 4) | (lane & 15);   // global output column
    const int mhi = (lane & 16) ? 8 : 0;         // C-fragment M offset
    const v8f z8  = {0.f, 0.f, 0.f, 0.f, 0.f, 0.f, 0.f, 0.f};

    // -------- GEMM1: e_in(64x256) @ We1; B held across M-tiles --------
    v8f acc1[MT] = {z8, z8, z8, z8};
#pragma unroll
    for (int kc = 0; kc < 8; ++kc) {
        v16h b = load_b_frag(we1h, 8, wave, kc, lane);
#pragma unroll
        for (int mt = 0; mt < MT; ++mt) {
            v16h a = load_a_lds(ein + mt * (16 * 256), 256, kc, lane);
            acc1[mt] = wmma_f16(a, b, acc1[mt]);
        }
    }
    {
        float bias1 = be1[n];
        float wlast = We1[(size_t)256 * DD + n];  // radial (K=256) rank-1 term, f32-exact
#pragma unroll
        for (int mt = 0; mt < MT; ++mt)
#pragma unroll
            for (int r = 0; r < 8; ++r) {
                int M = mt * 16 + r + mhi;
                float x = acc1[mt][r] + bias1 + rad_s[M] * wlast;
                m1s[M * 128 + n] = (_Float16)siluf_(x);
            }
    }
    __syncthreads();

    // -------- GEMM2: m1(64x128) @ We2 --------
    v8f acc2[MT] = {z8, z8, z8, z8};
#pragma unroll
    for (int kc = 0; kc < 4; ++kc) {
        v16h b = load_b_frag(we2h, 4, wave, kc, lane);
#pragma unroll
        for (int mt = 0; mt < MT; ++mt) {
            v16h a = load_a_lds(m1s + mt * (16 * 128), 128, kc, lane);
            acc2[mt] = wmma_f16(a, b, acc2[mt]);
        }
    }
    {
        float bias2 = be2[n];
        float wa_n  = Wa[n];
#pragma unroll
        for (int mt = 0; mt < MT; ++mt)
#pragma unroll
            for (int r = 0; r < 8; ++r) {
                int M = mt * 16 + r + mhi;
                float s = siluf_(acc2[mt][r] + bias2);
                m2s[M * 128 + n] = (_Float16)s;
                // attention logit partial: sum_n m[M][n]*Wa[n]
                float v = reduce16(s * wa_n);
                if ((lane & 15) == 0) atomicAdd(&satt[M], v);
            }
    }
    __syncthreads();

    // -------- attention gate + agg_h scatter (edge_feat = m * att) --------
    const float ba0 = ba[0];
#pragma unroll
    for (int mt = 0; mt < MT; ++mt)
#pragma unroll
        for (int r = 0; r < 8; ++r) {
            int M = mt * 16 + r + mhi;
            float att = sigmoidf_(satt[M] + ba0);
            float ef  = (float)m2s[M * 128 + n] * att;
            atomicAdd(&agg_h[(size_t)rows_s[M] * DD + n], ef);
        }

    // -------- GEMM3: edge_feat(64x128) @ Wc1 (A row-scaled by att at load) -----
    _Float16 attA[MT];
#pragma unroll
    for (int mt = 0; mt < MT; ++mt)
        attA[mt] = (_Float16)sigmoidf_(satt[mt * 16 + (lane & 15)] + ba0);

    v8f acc3[MT] = {z8, z8, z8, z8};
#pragma unroll
    for (int kc = 0; kc < 4; ++kc) {
        v16h b = load_b_frag(wc1h, 4, wave, kc, lane);
#pragma unroll
        for (int mt = 0; mt < MT; ++mt) {
            v16h a = load_a_lds(m2s + mt * (16 * 128), 128, kc, lane);
#pragma unroll
            for (int i = 0; i < 16; ++i) a[i] = a[i] * attA[mt];
            acc3[mt] = wmma_f16(a, b, acc3[mt]);
        }
    }
    {
        float bc1n = bc1[n];
        float wc2n = Wc2[n];
#pragma unroll
        for (int mt = 0; mt < MT; ++mt)
#pragma unroll
            for (int r = 0; r < 8; ++r) {
                int M = mt * 16 + r + mhi;
                float t = siluf_(acc3[mt][r] + bc1n);
                float v = reduce16(t * wc2n);
                if ((lane & 15) == 0) atomicAdd(&sw_s[M], v);
            }
    }
    __syncthreads();

    // -------- coord scatter: trans = coord_diff * w --------
    if (tid < EPB) {
        float wv = sw_s[tid];
        int   nd = rows_s[tid];
        atomicAdd(&agg_c[nd * 3 + 0], diff_s[tid][0] * wv);
        atomicAdd(&agg_c[nd * 3 + 1], diff_s[tid][1] * wv);
        atomicAdd(&agg_c[nd * 3 + 2], diff_s[tid][2] * wv);
        atomicAdd(&cnt[nd], 1.0f);
    }
}

// ---------------------------------------------------------------------------
// Node kernel: h_out = h + silu([h|agg_h]@Wn1+b)@Wn2+b ; coord mean update
// ---------------------------------------------------------------------------
__global__ __launch_bounds__(256) void egnn_node_kernel(
    const float* __restrict__ h, const _Float16* __restrict__ hf16,
    const float* __restrict__ coord,
    const float* __restrict__ agg_h, const float* __restrict__ agg_c,
    const float* __restrict__ cnt,
    const float* __restrict__ bn1, const float* __restrict__ bn2,
    const _Float16* __restrict__ wn1h, const _Float16* __restrict__ wn2h,
    float* __restrict__ out)
{
    __shared__ __align__(16) _Float16 nin[16 * 256];
    __shared__ __align__(16) _Float16 t1s[16 * 128];

    const int tid  = threadIdx.x;
    const int lane = tid & 31;
    const int wave = tid >> 5;
    const int n0   = blockIdx.x * 16;

    if (tid == 0) __builtin_prefetch(wn1h, 0, 1);

    // gather [h | agg_h] as f16
    {
        int m  = tid >> 4;
        int cb = (tid & 15) << 3;
        int node = n0 + m;
        *(U128*)(nin + m * 256 + cb) =
            *(const U128*)(hf16 + (size_t)node * DD + cb);
        const float4* pa = (const float4*)(agg_h + (size_t)node * DD + cb);
        float4 a = pa[0], b = pa[1];
        _Float16 t[8] = {(_Float16)a.x, (_Float16)a.y, (_Float16)a.z, (_Float16)a.w,
                         (_Float16)b.x, (_Float16)b.y, (_Float16)b.z, (_Float16)b.w};
        __builtin_memcpy(nin + m * 256 + 128 + cb, t, 16);
    }
    __syncthreads();

    const int n   = (wave << 4) | (lane & 15);
    const int mhi = (lane & 16) ? 8 : 0;

    v8f acc1 = {0.f, 0.f, 0.f, 0.f, 0.f, 0.f, 0.f, 0.f};
#pragma unroll
    for (int kc = 0; kc < 8; ++kc) {
        v16h a = load_a_lds(nin, 256, kc, lane);
        v16h b = load_b_frag(wn1h, 8, wave, kc, lane);
        acc1 = wmma_f16(a, b, acc1);
    }
    {
        float b1 = bn1[n];
#pragma unroll
        for (int r = 0; r < 8; ++r) {
            int M = r + mhi;
            t1s[M * 128 + n] = (_Float16)siluf_(acc1[r] + b1);
        }
    }
    __syncthreads();

    v8f acc2 = {0.f, 0.f, 0.f, 0.f, 0.f, 0.f, 0.f, 0.f};
#pragma unroll
    for (int kc = 0; kc < 4; ++kc) {
        v16h a = load_a_lds(t1s, 128, kc, lane);
        v16h b = load_b_frag(wn2h, 4, wave, kc, lane);
        acc2 = wmma_f16(a, b, acc2);
    }
    {
        float b2 = bn2[n];
#pragma unroll
        for (int r = 0; r < 8; ++r) {
            int M = r + mhi;
            size_t idx = (size_t)(n0 + M) * DD + n;
            out[idx] = acc2[r] + b2 + h[idx];   // residual (f32-exact)
        }
    }

    // coord_out = coord + agg_c / max(cnt, 1)
    if (tid < 16) {
        int node = n0 + tid;
        float inv = 1.0f / fmaxf(cnt[node], 1.0f);
#pragma unroll
        for (int c = 0; c < 3; ++c) {
            out[(size_t)NN * DD + node * 3 + c] =
                coord[node * 3 + c] + agg_c[node * 3 + c] * inv;
        }
    }
}

// ---------------------------------------------------------------------------
// Host launcher
// ---------------------------------------------------------------------------
extern "C" void kernel_launch(void* const* d_in, const int* in_sizes, int n_in,
                              void* d_out, int out_size, void* d_ws, size_t ws_size,
                              hipStream_t stream) {
    const float* h     = (const float*)d_in[0];
    const int*   eidx  = (const int*)d_in[1];
    const float* coord = (const float*)d_in[2];
    const float* We1   = (const float*)d_in[3];
    const float* be1   = (const float*)d_in[4];
    const float* We2   = (const float*)d_in[5];
    const float* be2   = (const float*)d_in[6];
    const float* Wa    = (const float*)d_in[7];
    const float* ba    = (const float*)d_in[8];
    const float* Wn1   = (const float*)d_in[9];
    const float* bn1   = (const float*)d_in[10];
    const float* Wn2   = (const float*)d_in[11];
    const float* bn2   = (const float*)d_in[12];
    const float* Wc1   = (const float*)d_in[13];
    const float* bc1   = (const float*)d_in[14];
    const float* Wc2   = (const float*)d_in[15];

    const int* erow = eidx;
    const int* ecol = eidx + EE;

    // workspace layout (bytes)
    constexpr size_t OFF_AGG_H = 0;
    constexpr size_t OFF_AGG_C = OFF_AGG_H + (size_t)NN * DD * 4;
    constexpr size_t OFF_CNT   = OFF_AGG_C + (size_t)NN * 3 * 4;
    constexpr size_t OFF_WE1H  = OFF_CNT + (size_t)NN * 4;        // 16B-aligned
    constexpr size_t SZ_F8 = 8 * 8 * 512 * 2;                     // K=256 weight frags
    constexpr size_t SZ_F4 = 8 * 4 * 512 * 2;                     // K=128 weight frags
    constexpr size_t OFF_WE2H = OFF_WE1H + SZ_F8;
    constexpr size_t OFF_WC1H = OFF_WE2H + SZ_F4;
    constexpr size_t OFF_WN1H = OFF_WC1H + SZ_F4;
    constexpr size_t OFF_WN2H = OFF_WN1H + SZ_F8;
    constexpr size_t OFF_HF16 = OFF_WN2H + SZ_F4;                 // f16 copy of h

    char* ws = (char*)d_ws;
    float*     agg_h = (float*)(ws + OFF_AGG_H);
    float*     agg_c = (float*)(ws + OFF_AGG_C);
    float*     cnt   = (float*)(ws + OFF_CNT);
    _Float16*  we1h  = (_Float16*)(ws + OFF_WE1H);
    _Float16*  we2h  = (_Float16*)(ws + OFF_WE2H);
    _Float16*  wc1h  = (_Float16*)(ws + OFF_WC1H);
    _Float16*  wn1h  = (_Float16*)(ws + OFF_WN1H);
    _Float16*  wn2h  = (_Float16*)(ws + OFF_WN2H);
    _Float16*  hf16  = (_Float16*)(ws + OFF_HF16);

    // zero accumulators (agg_h | agg_c | cnt are contiguous floats)
    const int zTot = NN * DD + NN * 3 + NN;
    zero_kernel<<<(zTot + 255) / 256, 256, 0, stream>>>(agg_h, zTot);

    // f16 copy of h for the gather path (halves gather bytes, kills per-edge cvt)
    const int hTot = NN * DD;
    cvt_f16_kernel<<<(hTot + 255) / 256, 256, 0, stream>>>(h, hf16, hTot);

    // pre-swizzle weights into f16 B-fragment order
    prep_frag_kernel<<<(8 * 8 * 512 + 255) / 256, 256, 0, stream>>>(We1, we1h, 8);
    prep_frag_kernel<<<(8 * 4 * 512 + 255) / 256, 256, 0, stream>>>(We2, we2h, 4);
    prep_frag_kernel<<<(8 * 4 * 512 + 255) / 256, 256, 0, stream>>>(Wc1, wc1h, 4);
    prep_frag_kernel<<<(8 * 8 * 512 + 255) / 256, 256, 0, stream>>>(Wn1, wn1h, 8);
    prep_frag_kernel<<<(8 * 4 * 512 + 255) / 256, 256, 0, stream>>>(Wn2, wn2h, 4);

    // edge pipeline: 320000 / 64 = 5000 blocks
    egnn_edge_kernel<<<EE / EPB, 256, 0, stream>>>(
        hf16, erow, ecol, coord, We1, be1, be2, Wa, ba, bc1, Wc2,
        we1h, we2h, wc1h, agg_h, agg_c, cnt);

    // node pipeline: 10000 / 16 = 625 blocks
    egnn_node_kernel<<<NN / 16, 256, 0, stream>>>(
        h, hf16, coord, agg_h, agg_c, cnt, bn1, bn2, wn1h, wn2h, (float*)d_out);
}
